// GNN_13761075216951
// MI455X (gfx1250) — matrix operand order repaired and checked by reading, not compiled, not run on hardware
//
#include <hip/hip_runtime.h>

typedef __attribute__((ext_vector_type(16))) _Float16 v16h;
typedef __attribute__((ext_vector_type(8)))  float    v8f;

#if __has_builtin(__builtin_amdgcn_global_load_async_to_lds_b128)
#define HAVE_ASYNC_LDS 1
// Types per hipcc diagnostic: param0 is "int __vector(4) __device__ *"
typedef int v4i_async __attribute__((__vector_size__(4 * sizeof(int))));
typedef __attribute__((address_space(1))) v4i_async* g_v4i_ptr;
typedef __attribute__((address_space(3))) v4i_async* l_v4i_ptr;
#endif

__device__ __forceinline__ void wait_async0() {
#ifdef HAVE_ASYNC_LDS
#if __has_builtin(__builtin_amdgcn_s_wait_asynccnt)
  __builtin_amdgcn_s_wait_asynccnt(0);
#else
  asm volatile("s_wait_asynccnt 0x0" ::: "memory");
#endif
#endif
}

// ---------------------------------------------------------------------------
// Degree / normalization precompute (shared by all 3 GCN layers)
// ---------------------------------------------------------------------------
__global__ void k_deg_init(float* deg, int n) {
  int i = blockIdx.x * blockDim.x + threadIdx.x;
  if (i < n) deg[i] = 1.0f;  // self-loop weight 1
}

__global__ void k_deg_edges(const int* __restrict__ dst, const float* __restrict__ ew,
                            float* deg, int E, int n) {
  int e = blockIdx.x * blockDim.x + threadIdx.x;
  if (e < E) {
    int d = dst[e];
    if ((unsigned)d < (unsigned)n) atomicAdd(&deg[d], ew[e]);
  }
}

__global__ void k_rsqrt(float* deg, int n) {
  int i = blockIdx.x * blockDim.x + threadIdx.x;
  if (i < n) {
    float v = deg[i];
    deg[i] = (v > 0.0f) ? rsqrtf(v) : 0.0f;
  }
}

__global__ void k_norm(const int* __restrict__ src, const int* __restrict__ dst,
                       const float* __restrict__ ew, const float* __restrict__ dinv,
                       float* __restrict__ norm, int E, int n) {
  int e = blockIdx.x * blockDim.x + threadIdx.x;
  if (e < E) {
    int s = src[e], d = dst[e];
    float a = ((unsigned)s < (unsigned)n) ? dinv[s] : 0.0f;
    float b = ((unsigned)d < (unsigned)n) ? dinv[d] : 0.0f;
    norm[e] = a * ew[e] * b;
  }
}

// ---------------------------------------------------------------------------
// One-time W pre-swizzle: f32 [K,N] row-major -> f16 blob in the exact CDNA5
// 16-bit B-fragment layout, grouped per 32-row K-step:
//   pos = ks*N*32 + (tile*32 + lane)*16 + half
//   lane = (n%16) + 16*((krel/8)%2), half = krel%8 + 8*(krel>=16), tile = n/16
// ---------------------------------------------------------------------------
template<int KDIM, int NDIM>
__global__ void k_preswz(const float* __restrict__ W, _Float16* __restrict__ out) {
  int i = blockIdx.x * blockDim.x + threadIdx.x;
  if (i < KDIM * NDIM) {
    int k = i / NDIM, n = i - (i / NDIM) * NDIM;
    int ks = k >> 5, krel = k & 31;
    int lane = (n & 15) + ((krel & 8) << 1);
    int half = (krel & 7) + ((krel & 16) >> 1);
    size_t pos = (size_t)ks * NDIM * 32 + (((n >> 4) * 32 + lane) << 4) + half;
    out[pos] = (_Float16)W[i];
  }
}

// ---------------------------------------------------------------------------
// WMMA GEMM: H[M,N] = (RELU? relu(X) : X)[M,K] * Wswz, f16 WMMA, f32 acc.
// Block = 256 threads = 8 waves; each wave does a 16-row stripe x N cols.
// Pre-swizzled W tile copied to LDS per K-step, double-buffered; async
// global->LDS (ASYNCcnt) when the builtin is available.
// ---------------------------------------------------------------------------
template<int KDIM, int NDIM, bool RELU>
__global__ __launch_bounds__(256) void k_gemm_wmma(const float* __restrict__ X,
                                                   const _Float16* __restrict__ Wswz,
                                                   float* __restrict__ H, int M) {
  constexpr int NT = NDIM / 16;   // 16x16 output tiles per wave
  constexpr int S  = KDIM / 32;   // K-steps
  __shared__ __align__(32) uint4 sBq[2][NDIM * 4];   // NDIM*64 bytes per buffer

  const int tid  = threadIdx.x;
  const int wave = tid >> 5;
  const int lane = tid & 31;
  const int m0   = blockIdx.x * 128 + wave * 16;
  const int rowA = min(m0 + (lane & 15), M - 1);      // clamp; stores guarded
  const int kbase = (lane < 16) ? 0 : 8;              // ISA 16-bit A layout

  auto stageW = [&](int buf, int ks) {
    const uint4* wsrc = (const uint4*)Wswz + (size_t)ks * NDIM * 4;
    #pragma unroll
    for (int idx = tid; idx < NDIM * 4; idx += 256) {
#ifdef HAVE_ASYNC_LDS
      g_v4i_ptr g = (g_v4i_ptr)(reinterpret_cast<uintptr_t>(wsrc + idx));
      l_v4i_ptr l = (l_v4i_ptr)((unsigned)(uintptr_t)&sBq[buf][idx]);
      __builtin_amdgcn_global_load_async_to_lds_b128(g, l, 0, 0);
#else
      sBq[buf][idx] = wsrc[idx];
#endif
    }
  };

  v8f acc[NT] = {};

  stageW(0, 0);
  wait_async0();
  __syncthreads();

  for (int ks = 0; ks < S; ++ks) {
    const int cur = ks & 1;
    if (ks + 1 < S) stageW(cur ^ 1, ks + 1);   // overlap copy with compute

    // A fragment per ISA 16-bit 16x32 layout: halves 0..7 = K kbase..kbase+7,
    // halves 8..15 = K kbase+16..kbase+23.
    const int k0 = ks * 32;
    const float* xp = X + (size_t)rowA * KDIM + k0 + kbase;
    if (ks + 1 < S) __builtin_prefetch(xp + 32, 0, 1);   // global_prefetch_b8
    float4 f0 = *(const float4*)(xp);
    float4 f1 = *(const float4*)(xp + 4);
    float4 f2 = *(const float4*)(xp + 16);
    float4 f3 = *(const float4*)(xp + 20);
    if (RELU) {
      f0.x = fmaxf(f0.x, 0.f); f0.y = fmaxf(f0.y, 0.f); f0.z = fmaxf(f0.z, 0.f); f0.w = fmaxf(f0.w, 0.f);
      f1.x = fmaxf(f1.x, 0.f); f1.y = fmaxf(f1.y, 0.f); f1.z = fmaxf(f1.z, 0.f); f1.w = fmaxf(f1.w, 0.f);
      f2.x = fmaxf(f2.x, 0.f); f2.y = fmaxf(f2.y, 0.f); f2.z = fmaxf(f2.z, 0.f); f2.w = fmaxf(f2.w, 0.f);
      f3.x = fmaxf(f3.x, 0.f); f3.y = fmaxf(f3.y, 0.f); f3.z = fmaxf(f3.z, 0.f); f3.w = fmaxf(f3.w, 0.f);
    }
    v16h a;
    a[0]  = (_Float16)f0.x; a[1]  = (_Float16)f0.y; a[2]  = (_Float16)f0.z; a[3]  = (_Float16)f0.w;
    a[4]  = (_Float16)f1.x; a[5]  = (_Float16)f1.y; a[6]  = (_Float16)f1.z; a[7]  = (_Float16)f1.w;
    a[8]  = (_Float16)f2.x; a[9]  = (_Float16)f2.y; a[10] = (_Float16)f2.z; a[11] = (_Float16)f2.w;
    a[12] = (_Float16)f3.x; a[13] = (_Float16)f3.y; a[14] = (_Float16)f3.z; a[15] = (_Float16)f3.w;

    const v16h* bFrag = (const v16h*)sBq[cur];
    #pragma unroll
    for (int t = 0; t < NT; ++t) {
      const v16h b = bFrag[t * 32 + lane];
      acc[t] = __builtin_amdgcn_wmma_f32_16x16x32_f16(
          false, a, false, b, (short)0, acc[t], false, false);
    }

    wait_async0();     // next buffer resident before anyone flips to it
    __syncthreads();   // all waves done reading cur before it is re-staged
  }

  // C/D layout: VGPR r, lane L -> row = r + 8*(L>=16), col = L%16
  const int rbase = (lane >= 16) ? 8 : 0;
  const int col   = lane & 15;
  #pragma unroll
  for (int t = 0; t < NT; ++t) {
    #pragma unroll
    for (int r = 0; r < 8; ++r) {
      int m = m0 + r + rbase;
      if (m < M) H[(size_t)m * NDIM + t * 16 + col] = acc[t][r];
    }
  }
}

// ---------------------------------------------------------------------------
// Aggregation: out = dinv^2 * H (self loop) + bias, then edge atomic scatter
// ---------------------------------------------------------------------------
template<int F>
__global__ void k_selfinit(const float* __restrict__ H, const float* __restrict__ dinv,
                           const float* __restrict__ bias, float* __restrict__ out, int n) {
  int i = blockIdx.x * blockDim.x + threadIdx.x;
  if (i < n * F) {
    int node = i / F;
    int f    = i - node * F;
    float dv = dinv[node];
    out[i] = H[i] * dv * dv + bias[f];
  }
}

template<int F>
__global__ void k_scatter(const int* __restrict__ src, const int* __restrict__ dst,
                          const float* __restrict__ norm, const float* __restrict__ H,
                          float* __restrict__ out, int E, int n) {
  constexpr int LPE = F / 4;                       // lanes per edge
  unsigned t = blockIdx.x * blockDim.x + threadIdx.x;
  unsigned e = t / LPE;
  int fo = (t - e * LPE) * 4;
  if (e < (unsigned)E) {
    int s = src[e], d = dst[e];
    if ((unsigned)s < (unsigned)n && (unsigned)d < (unsigned)n) {
      float w = norm[e];
      const float4 hv = *(const float4*)&H[(size_t)s * F + fo];
      float* o = &out[(size_t)d * F + fo];
      atomicAdd(o + 0, hv.x * w);
      atomicAdd(o + 1, hv.y * w);
      atomicAdd(o + 2, hv.z * w);
      atomicAdd(o + 3, hv.w * w);
    }
  }
}

// Layer 3: GEMV 64 -> 1 with ReLU on input (tiny; VALU is right here)
__global__ void k_gemv_relu(const float* __restrict__ X, const float* __restrict__ W,
                            float* __restrict__ h3, int M) {
  int i = blockIdx.x * blockDim.x + threadIdx.x;
  if (i < M) {
    const float* xp = X + (size_t)i * 64;
    float acc = 0.0f;
    #pragma unroll
    for (int k = 0; k < 64; k += 4) {
      float4 v = *(const float4*)(xp + k);
      acc += fmaxf(v.x, 0.f) * W[k + 0];
      acc += fmaxf(v.y, 0.f) * W[k + 1];
      acc += fmaxf(v.z, 0.f) * W[k + 2];
      acc += fmaxf(v.w, 0.f) * W[k + 3];
    }
    h3[i] = acc;
  }
}

__global__ void k_final_self(const float* __restrict__ h3, const float* __restrict__ dinv,
                             const float* __restrict__ b3, float* __restrict__ out, int n) {
  int i = blockIdx.x * blockDim.x + threadIdx.x;
  if (i < n) {
    float dv = dinv[i];
    out[i] = h3[i] * dv * dv + b3[0];
  }
}

__global__ void k_final_scatter(const int* __restrict__ src, const int* __restrict__ dst,
                                const float* __restrict__ norm, const float* __restrict__ h3,
                                float* __restrict__ out, int E, int n) {
  int e = blockIdx.x * blockDim.x + threadIdx.x;
  if (e < E) {
    int s = src[e], d = dst[e];
    if ((unsigned)s < (unsigned)n && (unsigned)d < (unsigned)n)
      atomicAdd(&out[d], h3[s] * norm[e]);
  }
}

// ---------------------------------------------------------------------------
extern "C" void kernel_launch(void* const* d_in, const int* in_sizes, int n_in,
                              void* d_out, int out_size, void* d_ws, size_t ws_size,
                              hipStream_t stream) {
  const float* x   = (const float*)d_in[0];
  const int*  eidx = (const int*)d_in[1];
  const float* ew  = (const float*)d_in[2];
  const float* W1  = (const float*)d_in[3];
  const float* b1  = (const float*)d_in[4];
  const float* W2  = (const float*)d_in[5];
  const float* b2  = (const float*)d_in[6];
  const float* W3  = (const float*)d_in[7];
  const float* b3  = (const float*)d_in[8];

  const int E  = in_sizes[2];
  const int Nn = in_sizes[0] / 512;
  const int* src = eidx;
  const int* dst = eidx + E;

  // workspace layout (float units), 256-element aligned chunks
  float* ws   = (float*)d_ws;
  size_t off  = 0;
  float* dinv = ws + off; off += (size_t)((Nn + 255) & ~255);
  float* norm = ws + off; off += (size_t)((E + 255) & ~255);
  float* Hbuf = ws + off; off += (size_t)Nn * 128;
  float* Xbuf = ws + off; off += (size_t)Nn * 128;
  float* h3   = ws + off; off += (size_t)((Nn + 255) & ~255);
  _Float16* Wswz1 = (_Float16*)(ws + off); off += (512 * 128) / 2;
  _Float16* Wswz2 = (_Float16*)(ws + off); off += (128 * 64) / 2;

  const int B = 256;
  auto cdiv = [](long long a, long long b) { return (unsigned)((a + b - 1) / b); };

  // --- normalization (once, reused by all 3 layers) + W pre-swizzle ---
  k_deg_init<<<cdiv(Nn, B), B, 0, stream>>>(dinv, Nn);
  k_deg_edges<<<cdiv(E, B), B, 0, stream>>>(dst, ew, dinv, E, Nn);
  k_rsqrt<<<cdiv(Nn, B), B, 0, stream>>>(dinv, Nn);
  k_norm<<<cdiv(E, B), B, 0, stream>>>(src, dst, ew, dinv, norm, E, Nn);
  k_preswz<512, 128><<<cdiv(512 * 128, B), B, 0, stream>>>(W1, Wswz1);
  k_preswz<128, 64><<<cdiv(128 * 64, B), B, 0, stream>>>(W2, Wswz2);

  const unsigned gemmGrid = cdiv(Nn, 128);

  // --- layer 1: 512 -> 128 ---
  k_gemm_wmma<512, 128, false><<<gemmGrid, 256, 0, stream>>>(x, Wswz1, Hbuf, Nn);
  k_selfinit<128><<<cdiv((long long)Nn * 128, B), B, 0, stream>>>(Hbuf, dinv, b1, Xbuf, Nn);
  k_scatter<128><<<cdiv((long long)E * 32, B), B, 0, stream>>>(src, dst, norm, Hbuf, Xbuf, E, Nn);

  // --- layer 2: 128 -> 64 (ReLU folded into A-fragment conversion) ---
  k_gemm_wmma<128, 64, true><<<gemmGrid, 256, 0, stream>>>(Xbuf, Wswz2, Hbuf, Nn);
  k_selfinit<64><<<cdiv((long long)Nn * 64, B), B, 0, stream>>>(Hbuf, dinv, b2, Xbuf, Nn);
  k_scatter<64><<<cdiv((long long)E * 16, B), B, 0, stream>>>(src, dst, norm, Hbuf, Xbuf, E, Nn);

  // --- layer 3: 64 -> 1 (GEMV) ---
  k_gemv_relu<<<cdiv(Nn, B), B, 0, stream>>>(Xbuf, W3, h3, Nn);
  k_final_self<<<cdiv(Nn, B), B, 0, stream>>>(h3, dinv, b3, (float*)d_out, Nn);
  k_final_scatter<<<cdiv(E, B), B, 0, stream>>>(src, dst, norm, h3, (float*)d_out, E, Nn);
}
